// SpatialSelfAttentionWin_25632364822690
// MI455X (gfx1250) — compile-verified, matrix operand order checked by reading, not run on hardware
//
#include <hip/hip_runtime.h>
#include <hip/hip_bf16.h>

typedef __attribute__((ext_vector_type(16))) _Float16 v16h;
typedef __attribute__((ext_vector_type(8)))  _Float16 v8h;
typedef __attribute__((ext_vector_type(8)))  float    v8f;
typedef __attribute__((ext_vector_type(4)))  int      v4i;

#define NHEAD 8

// gfx1250 async global->LDS path (ASYNCcnt-tracked), with sync fallback.
#if defined(__has_builtin)
#  if __has_builtin(__builtin_amdgcn_global_load_async_to_lds_b128) && \
      __has_builtin(__builtin_amdgcn_s_wait_asynccnt)
#    define USE_ASYNC_LDS 1
#  endif
#endif
#ifndef USE_ASYNC_LDS
#  define USE_ASYNC_LDS 0
#endif

#if USE_ASYNC_LDS
// Param 1: AS1 (global) v4i*, param 2: AS3 (LDS) v4i*, then imm offset, imm cpol.
#define ASYNC_CP_B128(lds_dst, glb_src)                                        \
    __builtin_amdgcn_global_load_async_to_lds_b128(                            \
        (__attribute__((address_space(1))) v4i*)(glb_src),                     \
        (__attribute__((address_space(3))) v4i*)(lds_dst), 0, 0)
#endif

// Load a WMMA 16x32 f16 fragment where logical "row" (A-matrix row, or
// B-matrix column stored as a row of a row-major LDS array) = lane%16.
// Per CDNA5 ISA 16-bit A layout: lanes 0-15 hold K = {kb..kb+7, kb+16..kb+23},
// lanes 16-31 hold K = {kb+8..kb+15, kb+24..kb+31}. Two 16B LDS reads.
__device__ __forceinline__ v16h ld_frag(const _Float16* base, int row0, int ld, int kbase) {
    const int lane = threadIdx.x & 31;
    const int r    = lane & 15;
    const int hi   = lane >> 4;
    const _Float16* p = base + (row0 + r) * ld + kbase + hi * 8;
    v8h lo = *(const v8h*)(p);
    v8h up = *(const v8h*)(p + 16);
    v16h out;
#pragma unroll
    for (int i = 0; i < 8; ++i) { out[i] = lo[i]; out[8 + i] = up[i]; }
    return out;
}

__device__ __forceinline__ v8f wmma_f16(v16h a, v16h b, v8f c) {
    return __builtin_amdgcn_wmma_f32_16x16x32_f16(false, a, false, b, (short)0, c, false, false);
}

// ---------------------------------------------------------------------------
// Kernel 1: per-(window, head) cosine attention.
// grid = 1024 windows * 8 heads = 8192 blocks, 256 threads (8 waves).
// Writes f16 result X[pixel][c] with pixel=((b*128+h)*128+w), c=head*64+d.
// ---------------------------------------------------------------------------
__global__ __launch_bounds__(256)
void win_attn_kernel(const float* __restrict__ qkv,
                     const float* __restrict__ logit_scale,
                     const float* __restrict__ rpb_table,
                     const int*   __restrict__ rel_index,
                     _Float16*    __restrict__ Xout) {
    // LDS carve-up (63232 B total):
    //   [0      ,17408) qs  f32 [64][68]   -> reused as attn f32 [64][68]
    //   [17408  ,34816) ks  f32 [64][68]   -> reused as ph  f16 [64][72]
    //   [34816  ,44032) qh  f16 [64][72]
    //   [44032  ,53248) kh  f16 [64][72]
    //   [53248  ,62464) vt  f16 [64][72]  (V transposed: vt[d][token])
    //   [62464  ,62976) invn f32 [128]
    __shared__ __align__(16) char smem[63232];
    float*    qs   = (float*)smem;
    float*    ks   = (float*)(smem + 17408);
    float*    attn = qs;
    _Float16* qh   = (_Float16*)(smem + 34816);
    _Float16* kh   = (_Float16*)(smem + 44032);
    _Float16* vt   = (_Float16*)(smem + 53248);
    _Float16* ph   = (_Float16*)(smem + 17408);
    float*    invn = (float*)(smem + 62464);

    const int t  = threadIdx.x;
    const int hh = blockIdx.x & 7;       // head
    const int wi = blockIdx.x >> 3;      // window id 0..1023
    const int b  = wi >> 8;
    const int nh = (wi >> 4) & 15;
    const int nw = wi & 15;

    const long plane = 128L * 128L;
    const float* qbase = qkv + (((long)b * NHEAD + hh) * 192L) * plane
                             + (long)(nh * 8) * 128 + (nw * 8);

    // Stage q,k (f32) and v (f16, transposed) into LDS.
#pragma unroll 4
    for (int it = 0; it < 16; ++it) {
        int idx = it * 256 + t;
        int d = idx >> 6, a = idx & 63, ys = a >> 3, xs = a & 7;
        long sp = (long)ys * 128 + xs;
        float qv = qbase[(long)d * plane + sp];
        float kv = qbase[(long)(64 + d) * plane + sp];
        float vv = qbase[(long)(128 + d) * plane + sp];
        qs[a * 68 + d] = qv;
        ks[a * 68 + d] = kv;
        vt[d * 72 + a] = (_Float16)vv;
    }
    __syncthreads();

    // Row L2 norms for q (rows 0..63) and k (rows 64..127).
    if (t < 128) {
        int row = t & 63;
        const float* src = (t < 64) ? (qs + row * 68) : (ks + row * 68);
        float ss = 0.f;
#pragma unroll
        for (int d = 0; d < 64; ++d) { float x = src[d]; ss += x * x; }
        invn[t] = 1.0f / fmaxf(sqrtf(ss), 1e-12f);
    }
    __syncthreads();

    // Normalized f16 copies.
#pragma unroll 4
    for (int it = 0; it < 16; ++it) {
        int idx = it * 256 + t;
        int a = idx >> 6, d = idx & 63;
        qh[a * 72 + d] = (_Float16)(qs[a * 68 + d] * invn[a]);
        kh[a * 72 + d] = (_Float16)(ks[a * 68 + d] * invn[64 + a]);
    }
    __syncthreads();

    const float scale = __expf(fminf(logit_scale[hh], 4.6051702f)); // clamp log(100)
    const int wave = t >> 5;
    const int lane = t & 31, r = lane & 15, hi = lane >> 4;

    // S = q_n * k_n^T : 16 tiles of 16x16, 2 per wave, K=64 in two steps.
#pragma unroll
    for (int tt = 0; tt < 2; ++tt) {
        int tile = wave * 2 + tt;
        int i = tile >> 2, j = tile & 3;
        v8f c = {};
#pragma unroll
        for (int kb = 0; kb < 64; kb += 32) {
            v16h af = ld_frag(qh, 16 * i, 72, kb);
            v16h bf = ld_frag(kh, 16 * j, 72, kb);  // B col n = k-row 16j+n
            c = wmma_f16(af, bf, c);
        }
#pragma unroll
        for (int rr = 0; rr < 8; ++rr) {
            int m = 16 * i + rr + 8 * hi;   // query token
            int n = 16 * j + r;             // key token
            int ridx = rel_index[m * 64 + n];
            float bias = rpb_table[ridx * NHEAD + hh];
            attn[m * 68 + n] = scale * c[rr] + bias;
        }
    }
    __syncthreads();

    // Softmax over keys; probabilities as f16 in ph.
    if (t < 64) {
        float* row = attn + t * 68;
        float mx = -1e30f;
#pragma unroll
        for (int n = 0; n < 64; ++n) mx = fmaxf(mx, row[n]);
        float s = 0.f;
#pragma unroll
        for (int n = 0; n < 64; ++n) s += __expf(row[n] - mx);
        float inv = 1.0f / s;
#pragma unroll
        for (int n = 0; n < 64; ++n)
            ph[t * 72 + n] = (_Float16)(__expf(row[n] - mx) * inv);
    }
    __syncthreads();

    // O = P * V : B col n = column (16j+n) of V = row of vt.
#pragma unroll
    for (int tt = 0; tt < 2; ++tt) {
        int tile = wave * 2 + tt;
        int i = tile >> 2, j = tile & 3;
        v8f c = {};
#pragma unroll
        for (int kb = 0; kb < 64; kb += 32) {
            v16h af = ld_frag(ph, 16 * i, 72, kb);
            v16h bf = ld_frag(vt, 16 * j, 72, kb);
            c = wmma_f16(af, bf, c);
        }
#pragma unroll
        for (int rr = 0; rr < 8; ++rr) {
            int a = 16 * i + rr + 8 * hi;   // token
            int d = 16 * j + r;             // channel within head
            int h = nh * 8 + (a >> 3), w = nw * 8 + (a & 7);
            long p = ((long)b * 128 + h) * 128 + w;
            Xout[p * 512 + hh * 64 + d] = (_Float16)c[rr];
        }
    }
}

// ---------------------------------------------------------------------------
// Kernel 2: convert proj_w (512x512 f32) to f16.
// ---------------------------------------------------------------------------
__global__ __launch_bounds__(256)
void cvt_w_kernel(const float* __restrict__ w, _Float16* __restrict__ wh) {
    int i = blockIdx.x * 256 + threadIdx.x;
    if (i < 512 * 512) wh[i] = (_Float16)w[i];
}

// ---------------------------------------------------------------------------
// Kernel 3: projection GEMM Y(65536x512) = X(65536x512) * W^T + b.
// 64x64 output tile per block, K staged 64 at a time, double-buffered LDS
// filled with async global->LDS B128 copies (ASYNCcnt) when available.
// grid = 1024 row-tiles * 8 col-tiles = 8192 blocks, 256 threads (8 waves).
// ---------------------------------------------------------------------------
__global__ __launch_bounds__(256)
void proj_kernel(const _Float16* __restrict__ X, const _Float16* __restrict__ Wh,
                 const float* __restrict__ bias, float* __restrict__ out) {
    __shared__ __align__(16) _Float16 Xs[2][64 * 72];
    __shared__ __align__(16) _Float16 Ws[2][64 * 72];
    const int t = threadIdx.x;
    const long pBase = (long)(blockIdx.x >> 3) * 64;
    const int  oBase = (blockIdx.x & 7) * 64;
    const int wave = t >> 5, lane = t & 31, r = lane & 15, hi = lane >> 4;
    const int i  = wave >> 1;
    const int j0 = (wave & 1) * 2, j1 = j0 + 1;

    // Tile copy assignment: thread -> rows (t>>3) and (t>>3)+32, 8 halves at
    // column (t&7)*8. 64 rows x 64 halves per operand per stage.
    const int ldrow = t >> 3;
    const int ldcol = (t & 7) * 8;
    const _Float16* Xg0 = X  + (pBase + ldrow) * 512 + ldcol;
    const _Float16* Xg1 = Xg0 + 32 * 512;
    const _Float16* Wg0 = Wh + (long)(oBase + ldrow) * 512 + ldcol;
    const _Float16* Wg1 = Wg0 + 32 * 512;
    const int lo0 = ldrow * 72 + ldcol;
    const int lo1 = (ldrow + 32) * 72 + ldcol;

    v8f c0 = {}, c1 = {};

#if USE_ASYNC_LDS
    // Prologue: async-fill buffer 0 with K-chunk [0,64).
    ASYNC_CP_B128(&Xs[0][lo0], Xg0);
    ASYNC_CP_B128(&Xs[0][lo1], Xg1);
    ASYNC_CP_B128(&Ws[0][lo0], Wg0);
    ASYNC_CP_B128(&Ws[0][lo1], Wg1);
#pragma unroll
    for (int s = 0; s < 8; ++s) {
        const int cur = s & 1;
        __builtin_amdgcn_s_wait_asynccnt(0);   // this wave's copies landed
        __syncthreads();                       // everyone's copies visible;
                                               // prev stage's readers done
        if (s < 7) {
            const int kb = (s + 1) * 64;
            ASYNC_CP_B128(&Xs[cur ^ 1][lo0], Xg0 + kb);
            ASYNC_CP_B128(&Xs[cur ^ 1][lo1], Xg1 + kb);
            ASYNC_CP_B128(&Ws[cur ^ 1][lo0], Wg0 + kb);
            ASYNC_CP_B128(&Ws[cur ^ 1][lo1], Wg1 + kb);
        }
#pragma unroll
        for (int kk = 0; kk < 64; kk += 32) {
            v16h af = ld_frag(&Xs[cur][0], 16 * i, 72, kk);
            v16h b0 = ld_frag(&Ws[cur][0], 16 * j0, 72, kk);
            v16h b1 = ld_frag(&Ws[cur][0], 16 * j1, 72, kk);
            c0 = wmma_f16(af, b0, c0);
            c1 = wmma_f16(af, b1, c1);
        }
    }
#else
    // Sync fallback: single buffer, register-staged copies.
    for (int s = 0; s < 8; ++s) {
        const int kb = s * 64;
        v8h x0 = *(const v8h*)(Xg0 + kb);
        v8h x1 = *(const v8h*)(Xg1 + kb);
        v8h w0 = *(const v8h*)(Wg0 + kb);
        v8h w1 = *(const v8h*)(Wg1 + kb);
        __syncthreads();
        *(v8h*)&Xs[0][lo0] = x0;
        *(v8h*)&Xs[0][lo1] = x1;
        *(v8h*)&Ws[0][lo0] = w0;
        *(v8h*)&Ws[0][lo1] = w1;
        __syncthreads();
#pragma unroll
        for (int kk = 0; kk < 64; kk += 32) {
            v16h af = ld_frag(&Xs[0][0], 16 * i, 72, kk);
            v16h b0 = ld_frag(&Ws[0][0], 16 * j0, 72, kk);
            v16h b1 = ld_frag(&Ws[0][0], 16 * j1, 72, kk);
            c0 = wmma_f16(af, b0, c0);
            c1 = wmma_f16(af, b1, c1);
        }
    }
#endif

#pragma unroll
    for (int rr = 0; rr < 8; ++rr) {
        int m = 16 * i + rr + 8 * hi;
        long p = pBase + m;
        int bb = (int)(p >> 14);
        int hw = (int)(p & 16383);
        int o0 = oBase + 16 * j0 + r;
        int o1 = oBase + 16 * j1 + r;
        out[((long)bb * 512 + o0) * 16384 + hw] = c0[rr] + bias[o0];
        out[((long)bb * 512 + o1) * 16384 + hw] = c1[rr] + bias[o1];
    }
}

extern "C" void kernel_launch(void* const* d_in, const int* in_sizes, int n_in,
                              void* d_out, int out_size, void* d_ws, size_t ws_size,
                              hipStream_t stream) {
    (void)in_sizes; (void)n_in; (void)out_size; (void)ws_size;
    const float* qkv         = (const float*)d_in[0];
    const float* logit_scale = (const float*)d_in[1];
    const float* rpb_table   = (const float*)d_in[2];
    const float* proj_w      = (const float*)d_in[3];
    const float* proj_b      = (const float*)d_in[4];
    const int*   rel_index   = (const int*)d_in[5];
    float* out = (float*)d_out;

    // Workspace: [0, 512KB) f16 proj weights; [512KB, 512KB+64MB) f16 X.
    _Float16* Wh = (_Float16*)d_ws;
    _Float16* X  = (_Float16*)((char*)d_ws + 524288);

    cvt_w_kernel<<<1024, 256, 0, stream>>>(proj_w, Wh);
    win_attn_kernel<<<8192, 256, 0, stream>>>(qkv, logit_scale, rpb_table, rel_index, X);
    proj_kernel<<<8192, 256, 0, stream>>>(X, Wh, proj_b, out);
}
